// Pct_63093069578674
// MI455X (gfx1250) — compile-verified, matrix operand order checked by reading, not run on hardware
//
#include <hip/hip_runtime.h>
#include <hip/hip_bf16.h>

// ---- sizes (fixed by the reference) ----
static constexpr int Bsz = 16;
static constexpr int Nn  = 2048;
static constexpr int CinD = 64;
static constexpr int Cc  = 256;
static constexpr int Mtot = Bsz * Nn;      // 32768 points
static constexpr int Mtiles = Mtot / 16;   // 2048

typedef __attribute__((ext_vector_type(8)))  __bf16 v8bf;
typedef __attribute__((ext_vector_type(16))) __bf16 v16bf;
typedef __attribute__((ext_vector_type(8)))  float  v8f;
typedef __attribute__((ext_vector_type(4)))  unsigned int u32x4;
typedef __attribute__((ext_vector_type(8)))  unsigned int u32x8;

// ---------------------------------------------------------------------------
// Fragment loader for V_WMMA_F32_16X16X32_BF16 (symmetric A/B layout).
// A 16x32 (MxK): lane L holds row r=L&15; kb=(L>>4)*8; elements are
// K = kb..kb+7 (VGPR0-3) and K = kb+16..kb+23 (VGPR4-7).
// B 32x16 (KxN): identical pattern with r = column index.
// Source is row-major with leading dim `ld`; two aligned 16B loads per lane.
// ---------------------------------------------------------------------------
__device__ __forceinline__ v16bf load_frag(const __bf16* __restrict__ base,
                                           int row0, int ld, int k0, int lane) {
  const int r  = lane & 15;
  const int kb = (lane >> 4) << 3;
  const __bf16* p = base + (size_t)(row0 + r) * ld + (k0 + kb);
  v8bf lo = *(const v8bf*)(p);
  v8bf hi = *(const v8bf*)(p + 16);
  return __builtin_shufflevector(lo, hi, 0,1,2,3,4,5,6,7,8,9,10,11,12,13,14,15);
}

__device__ __forceinline__ v8f wmma_bf16(v16bf a, v16bf b, v8f c) {
  // (neg_a, A, neg_b, B, c_mod, C, reuse_a, reuse_b)
  return __builtin_amdgcn_wmma_f32_16x16x32_bf16(false, a, false, b, (short)0, c,
                                                 false, false);
}

// ---------------------------------------------------------------------------
// Tensor Data Mover: DMA a [rows x 32] bf16 tile (row stride K elements) from
// global into LDS at byte offset `lds_off`. D# per CDNA5 ISA:
//   group0: count=1, lds_addr, global_addr[56:0], type=2
//   group1: data_size=1(2B), tensor_dim0=32, tensor_dim1=rows, tile_dim0=32,
//           tile_dim1=rows, tensor_dim0_stride=K
//   groups 2/3: benign (higher dims unused: tile_dim3/4 = 0)
// All descriptor values are wave-uniform -> SGPR "s" constraints.
// IMPORTANT: the LDS base pointer is passed as an (unused) asm operand so the
// pointer escapes -- otherwise LLVM proves the never-stored LDS global cannot
// be modified by the asm and folds all ds_loads to undef (observed round 2).
// ---------------------------------------------------------------------------
__device__ __forceinline__ void tdm_load_w_slice(const __bf16* wsrc, int k0,
                                                 int K, int rows,
                                                 unsigned lds_off,
                                                 __bf16* lds_base) {
  unsigned long long ga =
      (unsigned long long)(const void*)wsrc + (unsigned long long)k0 * 2ull;
  u32x4 g0;
  g0[0] = 1u;                                                // count = 1
  g0[1] = lds_off;                                           // lds_addr (bytes)
  g0[2] = (unsigned)(ga & 0xffffffffull);
  g0[3] = (unsigned)((ga >> 32) & 0x01ffffffull) | (2u << 30);  // type = 2
  u32x8 g1;
  g1[0] = 1u << 16;                    // data_size = 2 bytes
  g1[1] = 32u << 16;                   // tensor_dim0[15:0] = 32
  g1[2] = ((unsigned)rows) << 16;      // tensor_dim1[15:0]
  g1[3] = 32u << 16;                   // tile_dim0 = 32
  g1[4] = (unsigned)rows;              // tile_dim1
  g1[5] = (unsigned)K;                 // tensor_dim0_stride[31:0]
  g1[6] = 0u;
  g1[7] = 0u;
  u32x4 g2; g2[0] = 1u; g2[1] = 1u; g2[2] = 0u; g2[3] = 0u;   // dim2/3 unused
  u32x4 g3; g3[0] = 0u; g3[1] = 1u << 16; g3[2] = 0u; g3[3] = 0u;
  asm volatile("tensor_load_to_lds %0, %1, %2, %3"
               :: "s"(g0), "s"(g1), "s"(g2), "s"(g3), "s"(lds_base)
               : "memory");
}

// ---------------------------------------------------------------------------
// prep: fp32 -> bf16 conversion, BN folding
// ---------------------------------------------------------------------------
__global__ void cvt_bf16_kernel(const float* __restrict__ src,
                                __bf16* __restrict__ dst, int n) {
  int i = blockIdx.x * blockDim.x + threadIdx.x;
  if (i < n) dst[i] = (__bf16)src[i];
}

__global__ void bn_prep_kernel(const float* __restrict__ g, const float* __restrict__ b,
                               const float* __restrict__ m, const float* __restrict__ v,
                               float* __restrict__ scale, float* __restrict__ shift) {
  int i = threadIdx.x;  // 256 channels
  float s = g[i] * rsqrtf(v[i] + 1e-5f);
  scale[i] = s;
  shift[i] = b[i] - m[i] * s;
}

// ---------------------------------------------------------------------------
// Generic WMMA GEMM: Y[M,Cout] = epilogue( X[M,K] @ W[Cout,K]^T )
// Block = 8 waves; the W k-slices [Cout x 32] are staged into LDS by the
// Tensor Data Mover, DOUBLE BUFFERED: wave 0 launches the DMA for slice k+1
// before the block computes on slice k, and waits tensorcnt only at the end
// of the iteration, overlapping DMA with WMMA work. All waves build
// B-fragments from LDS (ds_load). One wave computes a 16x64 output strip.
// mode 0: bn+relu -> bf16 row-major
// mode 1: bn+relu -> bf16 + f32 row-major
// mode 2: plain   -> bf16 row-major
// mode 3: +bias   -> bf16 transposed [B,C,N]
// mode 4: +bias, bn+relu, +h residual -> f32 transposed [B,C,N]
// ---------------------------------------------------------------------------
__global__ __launch_bounds__(256) void gemm_wmma_kernel(
    const __bf16* __restrict__ X, const __bf16* __restrict__ W,
    int K, int Cout,
    const float* __restrict__ scale, const float* __restrict__ shift,
    const float* __restrict__ bias, const float* __restrict__ hres,
    __bf16* __restrict__ obf, float* __restrict__ of32, int mode) {
  __shared__ __align__(16) __bf16 ldsW[2][256 * 32];  // 2 x 16 KB k-slices
  const int lane = threadIdx.x & 31;
  const int wid  = (blockIdx.x * blockDim.x + threadIdx.x) >> 5;
  const int nquads = Cout >> 6;
  const int mt = wid / nquads;      // grid sized exactly: mt < Mtiles always
  const int nq = wid % nquads;
  const int m0 = mt << 4;
  const int nb = nq << 6;
  const bool issuer = (threadIdx.x >> 5) == 0;

  // prologue: fill buffer 0
  if (issuer) {
    tdm_load_w_slice(W, 0, K, Cout, 0u, &ldsW[0][0]);
    __builtin_amdgcn_s_wait_tensorcnt(0);
  }
  __syncthreads();

  v8f vz = {};
  v8f acc[4] = {vz, vz, vz, vz};
  for (int k0 = 0; k0 < K; k0 += 32) {
    const int cb = (k0 >> 5) & 1;
    const bool more = (k0 + 32) < K;
    if (more && issuer)  // async: launch next slice into the other buffer
      tdm_load_w_slice(W, k0 + 32, K, Cout,
                       (unsigned)((1 - cb) * 256 * 32 * 2), &ldsW[0][0]);
    v16bf a = load_frag(X, m0, K, k0, lane);
    const __bf16* wb = &ldsW[cb][0];
#pragma unroll
    for (int t = 0; t < 4; ++t) {
      v16bf bw = load_frag(wb, nb + (t << 4), 32, 0, lane);
      acc[t] = wmma_bf16(a, bw, acc[t]);
    }
    if (more && issuer) __builtin_amdgcn_s_wait_tensorcnt(0);
    __syncthreads();  // next slice ready; this slice's reads already consumed
  }

  const int n  = lane & 15;
  const int mb = (lane >> 4) << 3;
#pragma unroll
  for (int t = 0; t < 4; ++t) {
    const int c = nb + (t << 4) + n;
    const float sc = scale ? scale[c] : 1.0f;
    const float sh = shift ? shift[c] : 0.0f;
    const float bi = bias  ? bias[c]  : 0.0f;
#pragma unroll
    for (int k = 0; k < 8; ++k) {
      const int row = m0 + mb + k;
      float y = acc[t][k] + bi;
      if (scale) y = fmaxf(y * sc + sh, 0.0f);
      if (mode <= 2) {
        obf[(size_t)row * Cout + c] = (__bf16)y;
        if (mode == 1) of32[(size_t)row * Cout + c] = y;
      } else if (mode == 3) {
        const int b = row >> 11, pn = row & (Nn - 1);
        obf[((size_t)b * Cout + c) * Nn + pn] = (__bf16)y;
      } else {  // mode 4: out = h + t, transposed
        y += hres[(size_t)row * Cout + c];
        const int b = row >> 11, pn = row & (Nn - 1);
        of32[((size_t)b * Cout + c) * Nn + pn] = y;
      }
    }
  }
}

// ---------------------------------------------------------------------------
// Attention pass 1: per (b, 16-row i-tile) wave, stream j tiles with WMMA,
// online softmax -> rowmax / rowsum. E tile C-layout: lane holds
// (m = k + 8*(lane>>4), n = lane&15); cross-lane combine over 16-lane groups.
// ---------------------------------------------------------------------------
__global__ __launch_bounds__(256) void attn_stats_kernel(
    const __bf16* __restrict__ q, float* __restrict__ rmax,
    float* __restrict__ rsum) {
  const int lane = threadIdx.x & 31;
  const int wid  = (blockIdx.x * blockDim.x + threadIdx.x) >> 5;
  const int b  = wid >> 7;          // N/16 = 128 i-tiles per batch
  const int i0 = (wid & 127) << 4;
  const __bf16* qb = q + (size_t)b * Nn * Cc;

  v16bf afr[8];
#pragma unroll
  for (int kc = 0; kc < 8; ++kc) afr[kc] = load_frag(qb, i0, Cc, kc * 32, lane);

  float m8[8], l8[8];
#pragma unroll
  for (int k = 0; k < 8; ++k) { m8[k] = -1e30f; l8[k] = 0.0f; }

  for (int j0 = 0; j0 < Nn; j0 += 16) {
    if (j0 + 16 < Nn)  // hide next tile's latency (global_prefetch_b8)
      __builtin_prefetch((const void*)(qb + (size_t)(j0 + 16 + (lane & 15)) * Cc), 0, 1);
    v8f e = {};
#pragma unroll
    for (int kc = 0; kc < 8; ++kc) {
      v16bf bf = load_frag(qb, j0, Cc, kc * 32, lane);  // q rows j as B columns
      e = wmma_bf16(afr[kc], bf, e);
    }
#pragma unroll
    for (int k = 0; k < 8; ++k) {
      float mn = fmaxf(m8[k], e[k]);
      l8[k] = l8[k] * __expf(m8[k] - mn) + __expf(e[k] - mn);
      m8[k] = mn;
    }
  }
  // combine across the 16 lanes sharing each row
#pragma unroll
  for (int off = 1; off < 16; off <<= 1) {
#pragma unroll
    for (int k = 0; k < 8; ++k) {
      float mo = __shfl_xor(m8[k], off, 32);
      float lo = __shfl_xor(l8[k], off, 32);
      float mn = fmaxf(m8[k], mo);
      l8[k] = l8[k] * __expf(m8[k] - mn) + lo * __expf(mo - mn);
      m8[k] = mn;
    }
  }
  if ((lane & 15) == 0) {
    int base = b * Nn + i0 + ((lane >> 4) << 3);
#pragma unroll
    for (int k = 0; k < 8; ++k) { rmax[base + k] = m8[k]; rsum[base + k] = l8[k]; }
  }
}

// ---------------------------------------------------------------------------
// Attention pass 2: per (b, 16-col j-tile) wave.
// Recompute E' = Q_j x Q_i^T (= P^T tile), apply exp/l_i, stage the 16x32
// bf16 P^T tile through wave-private LDS to get A-fragment layout, then
// accumulate x_r = P^T @ V over all 256 channels with 16 WMMAs / i-chunk.
// Q_j A-fragments are register-cached (reused 128x). Column sums give the
// L1 renorm; writes hmr = h - x_r (bf16).
// ---------------------------------------------------------------------------
__global__ __launch_bounds__(256) void attn_apply_kernel(
    const __bf16* __restrict__ q, const __bf16* __restrict__ vt,
    const float* __restrict__ rmax, const float* __restrict__ rsum,
    const float* __restrict__ hf, __bf16* __restrict__ hmr) {
  __shared__ __align__(16) __bf16 pt[8][16 * 32];  // per-wave P^T staging
  const int lane = threadIdx.x & 31;
  const int w    = threadIdx.x >> 5;
  const int wid  = (blockIdx.x * blockDim.x + threadIdx.x) >> 5;
  const int b  = wid >> 7;
  const int j0 = (wid & 127) << 4;
  const __bf16* qb  = q  + (size_t)b * Nn * Cc;
  const __bf16* vtb = vt + (size_t)b * Cc * Nn;  // v transposed [C,N]
  const int sb = b * Nn;
  const int n = lane & 15;
  const int g = lane >> 4;
  __bf16* myp = &pt[w][0];

  v16bf afr[8];  // Q rows j, cached across all 128 i-chunks
#pragma unroll
  for (int kc = 0; kc < 8; ++kc) afr[kc] = load_frag(qb, j0, Cc, kc * 32, lane);

  v8f vz = {};
  v8f acc[16];
#pragma unroll
  for (int t = 0; t < 16; ++t) acc[t] = vz;
  float ssum[8] = {0, 0, 0, 0, 0, 0, 0, 0};

  for (int i0 = 0; i0 < Nn; i0 += 32) {
    if (i0 + 32 < Nn) {  // prefetch next chunk's Q rows and V lanes
      __builtin_prefetch((const void*)(qb + (size_t)(i0 + 32 + n) * Cc), 0, 1);
      __builtin_prefetch((const void*)(vtb + (size_t)n * Nn + i0 + 32), 0, 1);
    }
#pragma unroll
    for (int half = 0; half < 2; ++half) {
      const int ic = i0 + (half << 4);
      v8f e = vz;
#pragma unroll
      for (int kc = 0; kc < 8; ++kc) {
        v16bf bb = load_frag(qb, ic, Cc, kc * 32, lane);  // rows i as cols
        e = wmma_bf16(afr[kc], bb, e);
      }
      // column of this tile is query-row i = ic + n -> softmax stats of row i
      const float mi  = rmax[sb + ic + n];
      const float inv = 1.0f / rsum[sb + ic + n];  // >= 1, safe
#pragma unroll
      for (int k = 0; k < 8; ++k) {
        float p = __expf(e[k] - mi) * inv;
        ssum[k] += p;                                     // partial colsum s_j
        myp[(k + (g << 3)) * 32 + (half << 4) + n] = (__bf16)p;
      }
    }
    asm volatile("s_wait_dscnt 0x0" ::: "memory");
    v16bf ap = load_frag(myp, 0, 32, 0, lane);  // P^T as A fragment (16x32)
#pragma unroll
    for (int ct = 0; ct < 16; ++ct) {
      v16bf bv = load_frag(vtb, ct << 4, Nn, i0, lane);  // V: K=i, N=channel
      acc[ct] = wmma_bf16(ap, bv, acc[ct]);
    }
    asm volatile("s_wait_dscnt 0x0" ::: "memory");
  }
  // reduce colsums over the 16 lanes sharing each output row
#pragma unroll
  for (int off = 1; off < 16; off <<= 1)
#pragma unroll
    for (int k = 0; k < 8; ++k) ssum[k] += __shfl_xor(ssum[k], off, 32);

#pragma unroll
  for (int ct = 0; ct < 16; ++ct) {
    const int c = (ct << 4) + n;
#pragma unroll
    for (int k = 0; k < 8; ++k) {
      const int row = j0 + (g << 3) + k;
      const size_t gp = (size_t)sb + row;
      float xr = acc[ct][k] / (1e-9f + ssum[k]);
      float d  = hf[gp * Cc + c] - xr;
      hmr[gp * Cc + c] = (__bf16)d;
    }
  }
}

// ---------------------------------------------------------------------------
extern "C" void kernel_launch(void* const* d_in, const int* in_sizes, int n_in,
                              void* d_out, int out_size, void* d_ws, size_t ws_size,
                              hipStream_t stream) {
  (void)in_sizes; (void)n_in; (void)out_size; (void)ws_size;
  const float* x    = (const float*)d_in[0];
  const float* w1   = (const float*)d_in[1];
  const float* w2   = (const float*)d_in[2];
  const float* wqk  = (const float*)d_in[3];
  const float* wv   = (const float*)d_in[4];
  const float* bv   = (const float*)d_in[5];
  const float* wt   = (const float*)d_in[6];
  const float* bt   = (const float*)d_in[7];
  const float* bn1g = (const float*)d_in[8],  *bn1b = (const float*)d_in[9];
  const float* bn1m = (const float*)d_in[10], *bn1v = (const float*)d_in[11];
  const float* bn2g = (const float*)d_in[12], *bn2b = (const float*)d_in[13];
  const float* bn2m = (const float*)d_in[14], *bn2v = (const float*)d_in[15];
  const float* bn3g = (const float*)d_in[16], *bn3b = (const float*)d_in[17];
  const float* bn3m = (const float*)d_in[18], *bn3v = (const float*)d_in[19];

  char* ws = (char*)d_ws;
  size_t off = 0;
  auto alloc = [&](size_t bytes) -> void* {
    off = (off + 255) & ~(size_t)255;
    void* p = ws + off;
    off += bytes;
    return p;
  };
  __bf16* xb   = (__bf16*)alloc((size_t)Mtot * CinD * 2);
  __bf16* w1b  = (__bf16*)alloc((size_t)Cc * CinD * 2);
  __bf16* w2b  = (__bf16*)alloc((size_t)Cc * Cc * 2);
  __bf16* wqkb = (__bf16*)alloc((size_t)Cc * Cc * 2);
  __bf16* wvb  = (__bf16*)alloc((size_t)Cc * Cc * 2);
  __bf16* wtb  = (__bf16*)alloc((size_t)Cc * Cc * 2);
  float* s1 = (float*)alloc(Cc * 4); float* t1 = (float*)alloc(Cc * 4);
  float* s2 = (float*)alloc(Cc * 4); float* t2 = (float*)alloc(Cc * 4);
  float* s3 = (float*)alloc(Cc * 4); float* t3 = (float*)alloc(Cc * 4);
  __bf16* h1b = (__bf16*)alloc((size_t)Mtot * Cc * 2);
  __bf16* hb  = (__bf16*)alloc((size_t)Mtot * Cc * 2);
  float*  hf  = (float*) alloc((size_t)Mtot * Cc * 4);
  __bf16* qb  = (__bf16*)alloc((size_t)Mtot * Cc * 2);
  __bf16* vtb = (__bf16*)alloc((size_t)Mtot * Cc * 2);
  __bf16* hmr = (__bf16*)alloc((size_t)Mtot * Cc * 2);
  float* rmx  = (float*)alloc((size_t)Mtot * 4);
  float* rsm  = (float*)alloc((size_t)Mtot * 4);

  auto cvt = [&](const float* s, __bf16* d, int nel) {
    cvt_bf16_kernel<<<(nel + 255) / 256, 256, 0, stream>>>(s, d, nel);
  };
  cvt(x, xb, Mtot * CinD);
  cvt(w1, w1b, Cc * CinD);
  cvt(w2, w2b, Cc * Cc);
  cvt(wqk, wqkb, Cc * Cc);
  cvt(wv, wvb, Cc * Cc);
  cvt(wt, wtb, Cc * Cc);
  bn_prep_kernel<<<1, 256, 0, stream>>>(bn1g, bn1b, bn1m, bn1v, s1, t1);
  bn_prep_kernel<<<1, 256, 0, stream>>>(bn2g, bn2b, bn2m, bn2v, s2, t2);
  bn_prep_kernel<<<1, 256, 0, stream>>>(bn3g, bn3b, bn3m, bn3v, s3, t3);

  const int gemm_blocks = (Mtiles * (Cc / 64) * 32) / 256;  // 1024
  // layer 1: h1 = relu(bn1(x @ w1^T))
  gemm_wmma_kernel<<<gemm_blocks, 256, 0, stream>>>(
      xb, w1b, CinD, Cc, s1, t1, nullptr, nullptr, h1b, nullptr, 0);
  // layer 2: h = relu(bn2(h1 @ w2^T))  (bf16 + fp32 copies)
  gemm_wmma_kernel<<<gemm_blocks, 256, 0, stream>>>(
      h1b, w2b, Cc, Cc, s2, t2, nullptr, nullptr, hb, hf, 1);
  // q = h @ wqk^T
  gemm_wmma_kernel<<<gemm_blocks, 256, 0, stream>>>(
      hb, wqkb, Cc, Cc, nullptr, nullptr, nullptr, nullptr, qb, nullptr, 2);
  // v = h @ wv^T + bv, stored transposed [B,C,N]
  gemm_wmma_kernel<<<gemm_blocks, 256, 0, stream>>>(
      hb, wvb, Cc, Cc, nullptr, nullptr, bv, nullptr, vtb, nullptr, 3);

  // attention (flash-style, 2048 waves each)
  attn_stats_kernel<<<256, 256, 0, stream>>>(qb, rmx, rsm);
  attn_apply_kernel<<<256, 256, 0, stream>>>(qb, vtb, rmx, rsm, hf, hmr);

  // out = transpose(h + relu(bn3((h - x_r) @ wt^T + bt)))
  gemm_wmma_kernel<<<gemm_blocks, 256, 0, stream>>>(
      hmr, wtb, Cc, Cc, s3, t3, bt, hf, nullptr, (float*)d_out, 4);
}
